// ContentOnlyPhasorBlock_37082747633862
// MI455X (gfx1250) — matrix operand order repaired and checked by reading, not compiled, not run on hardware
//
#include <hip/hip_runtime.h>
#include <math.h>

// ---------------------------------------------------------------------------
// CDNA5 (gfx1250) wave32 WMMA bf16 building blocks
// ---------------------------------------------------------------------------
typedef __bf16 bf16_t;
typedef __attribute__((ext_vector_type(16))) __bf16 v16bf;
typedef __attribute__((ext_vector_type(8)))  __bf16 v8bf;
typedef __attribute__((ext_vector_type(8)))  float  v8f;

union BF16Frag { v16bf v; v8bf h[2]; };

// A-matrix fragment (16x32 bf16), source row-major with leading dim ld.
// ISA 7.12.2: lanes 0-15 -> row M=lane, elems 0..7 = K k0..k0+7, elems 8..15 = k0+16..k0+23
//             lanes 16-31 -> row M=lane-16, K ranges shifted by +8.
__device__ __forceinline__ v16bf load_frag_a(const bf16_t* __restrict__ p, int ld,
                                             int row0, int k0, int lane) {
  const int r  = row0 + (lane & 15);
  const int kk = k0 + ((lane & 16) ? 8 : 0);
  BF16Frag f;
  f.h[0] = *reinterpret_cast<const v8bf*>(p + (size_t)r * ld + kk);
  f.h[1] = *reinterpret_cast<const v8bf*>(p + (size_t)r * ld + kk + 16);
  return f.v;
}

// B-matrix fragment (32x16 bf16) taken from a TRANSPOSED operand Bt[n][k]
// (row-major, leading dim ld): lanes 0-15 -> col N=lane, elems 0..7 = K k0..k0+7, elems 8..15 = k0+16..k0+23;
// lanes 16-31 -> col N=lane-16, K ranges shifted by +8.
__device__ __forceinline__ v16bf load_frag_b(const bf16_t* __restrict__ bt, int ld,
                                             int n0, int k0, int lane) {
  const int n  = n0 + (lane & 15);
  const int kk = k0 + ((lane & 16) ? 8 : 0);
  BF16Frag f;
  f.h[0] = *reinterpret_cast<const v8bf*>(bt + (size_t)n * ld + kk);
  f.h[1] = *reinterpret_cast<const v8bf*>(bt + (size_t)n * ld + kk + 16);
  return f.v;
}

__device__ __forceinline__ v8f wmma_bf16(v16bf a, v16bf b, v8f c) {
  return __builtin_amdgcn_wmma_f32_16x16x32_bf16(false, a, false, b, (short)0, c, false, false);
}

// ---------------------------------------------------------------------------
// Prep kernels: fp32 -> bf16 convert / transpose-convert
// ---------------------------------------------------------------------------
__global__ void cvt_f32_bf16(const float* __restrict__ in, bf16_t* __restrict__ out, int n) {
  int i = blockIdx.x * blockDim.x + threadIdx.x;
  if (i < n) out[i] = (bf16_t)in[i];
}

__global__ void transpose_f32_bf16(const float* __restrict__ in, bf16_t* __restrict__ out,
                                   int R, int C) {
  int i = blockIdx.x * blockDim.x + threadIdx.x;
  if (i < R * C) {
    int r = i / C, c = i % C;
    out[(size_t)c * R + r] = (bf16_t)in[i];
  }
}

// ---------------------------------------------------------------------------
// Generic fused GEMM: out = epilogue(X(MxK) @ Wt(NxK)^T + bias)
// Block = 256 threads = 8 waves; each wave computes a 16x64 output tile.
// The k-loop is software-pipelined: fragments for step k+32 are loaded while
// the WMMAs of step k issue, so the matrix pipe never waits on loadcnt==0.
// MODE: 0 -> bf16 store         1 -> exact GELU, bf16 store
//       2 -> +residual, f32     3 -> softplus()+0.1, f32
//       4 -> bf16 TRANSPOSED store (out[n*M + row])
// ---------------------------------------------------------------------------
template<int MODE>
__global__ void gemm_kernel(const bf16_t* __restrict__ X, const bf16_t* __restrict__ Wt,
                            const float* __restrict__ bias, const float* __restrict__ residual,
                            void* __restrict__ outp, int M, int N, int Kd) {
  const int lane = threadIdx.x & 31;
  const int wave = threadIdx.x >> 5;
  const int m0 = blockIdx.x * 128 + wave * 16;
  const int n0 = blockIdx.y * 64;

  v8f acc[4] = {};

  // prologue: stage k0 = 0
  v16bf a = load_frag_a(X, Kd, m0, 0, lane);
  v16bf b[4];
#pragma unroll
  for (int j = 0; j < 4; ++j) b[j] = load_frag_b(Wt, Kd, n0 + j * 16, 0, lane);

  for (int k0 = 32; k0 < Kd; k0 += 32) {
    // stage k0 while consuming k0-32
    v16bf an = load_frag_a(X, Kd, m0, k0, lane);
    v16bf bn[4];
#pragma unroll
    for (int j = 0; j < 4; ++j) bn[j] = load_frag_b(Wt, Kd, n0 + j * 16, k0, lane);
#pragma unroll
    for (int j = 0; j < 4; ++j) acc[j] = wmma_bf16(a, b[j], acc[j]);
    a = an;
#pragma unroll
    for (int j = 0; j < 4; ++j) b[j] = bn[j];
  }
#pragma unroll
  for (int j = 0; j < 4; ++j) acc[j] = wmma_bf16(a, b[j], acc[j]);   // tail

  const int hi = (lane & 16) ? 8 : 0;
  const int nc = lane & 15;
#pragma unroll
  for (int j = 0; j < 4; ++j) {
    const int n = n0 + j * 16 + nc;
#pragma unroll
    for (int i = 0; i < 8; ++i) {
      const int row = m0 + i + hi;
      float v = acc[j][i] + bias[n];
      if (MODE == 1) v = 0.5f * v * (1.0f + erff(v * 0.70710678118654752f));
      if (MODE == 3) { v = (v > 20.f) ? v : log1pf(expf(v)); v += 0.1f; }
      if (MODE == 2) {
        ((float*)outp)[(size_t)row * N + n] = v + residual[(size_t)row * N + n];
      } else if (MODE == 3) {
        ((float*)outp)[(size_t)row * N + n] = v;
      } else if (MODE == 4) {
        ((bf16_t*)outp)[(size_t)n * M + row] = (bf16_t)v;
      } else {
        ((bf16_t*)outp)[(size_t)row * N + n] = (bf16_t)v;
      }
    }
  }
}

// ---------------------------------------------------------------------------
// Phase head: ph = tanh(H @ W2t^T + b2) * pi ; QK[l, 0:64]=amp*cos, [64:128]=amp*sin
// Same software-pipelined k-loop as gemm_kernel (N fixed at 64).
// ---------------------------------------------------------------------------
__global__ void phase_kernel(const bf16_t* __restrict__ H, const bf16_t* __restrict__ W2t,
                             const float* __restrict__ b2, const float* __restrict__ amp,
                             bf16_t* __restrict__ QK, int Kd) {
  const int lane = threadIdx.x & 31;
  const int wave = threadIdx.x >> 5;
  const int m0 = blockIdx.x * 128 + wave * 16;

  v8f acc[4] = {};
  v16bf a = load_frag_a(H, Kd, m0, 0, lane);
  v16bf b[4];
#pragma unroll
  for (int j = 0; j < 4; ++j) b[j] = load_frag_b(W2t, Kd, j * 16, 0, lane);

  for (int k0 = 32; k0 < Kd; k0 += 32) {
    v16bf an = load_frag_a(H, Kd, m0, k0, lane);
    v16bf bn[4];
#pragma unroll
    for (int j = 0; j < 4; ++j) bn[j] = load_frag_b(W2t, Kd, j * 16, k0, lane);
#pragma unroll
    for (int j = 0; j < 4; ++j) acc[j] = wmma_bf16(a, b[j], acc[j]);
    a = an;
#pragma unroll
    for (int j = 0; j < 4; ++j) b[j] = bn[j];
  }
#pragma unroll
  for (int j = 0; j < 4; ++j) acc[j] = wmma_bf16(a, b[j], acc[j]);

  const int hi = (lane & 16) ? 8 : 0;
  const int nc = lane & 15;
  const float PI_F = 3.14159265358979323846f;
#pragma unroll
  for (int j = 0; j < 4; ++j) {
    const int n = j * 16 + nc;
#pragma unroll
    for (int i = 0; i < 8; ++i) {
      const int row = m0 + i + hi;
      float ph = tanhf(acc[j][i] + b2[n]) * PI_F;
      float am = amp[(size_t)row * 64 + n];
      QK[(size_t)row * 128 + n]      = (bf16_t)(am * cosf(ph));
      QK[(size_t)row * 128 + 64 + n] = (bf16_t)(am * sinf(ph));
    }
  }
}

// ---------------------------------------------------------------------------
// Causal linear attention: Ret = diag(1/sqrt((l+1)K)) * tril(Q Kc^T) @ V
// Q,Kc: (L,128) bf16 ; Vt: (D,L) bf16 (transposed) ; Ret: (L,D) f32.
// grid = (L/64, D/256), block = 256 threads (8 waves).
// Per key block: S (64x64) via WMMA -> mask -> bf16 in LDS -> acc += S@V via WMMA.
// Fragment loads are batched ahead of each WMMA group; next key block's K/V
// tiles are prefetched (global_prefetch_b8) a full block ahead.
// ---------------------------------------------------------------------------
__global__ void attention_kernel(const bf16_t* __restrict__ Q, const bf16_t* __restrict__ Kc,
                                 const bf16_t* __restrict__ Vt, float* __restrict__ Ret,
                                 int L, int D) {
  __shared__ bf16_t Sl[64 * 80];   // stride 80 keeps every b128 fragment load 16B-aligned

  const int lane = threadIdx.x & 31;
  const int wave = threadIdx.x >> 5;
  const int mb   = blockIdx.x;
  const int m0g  = mb * 64;
  const int dblk = blockIdx.y * 256;
  const int mt = wave >> 1;                // row tile (0..3) owned in P phase
  const int d0 = dblk + (wave & 1) * 128;  // 128-col half owned in P phase
  const int hi = (lane & 16) ? 8 : 0;
  const int nc = lane & 15;

  // this wave's two S subtiles (of the 16 in the 64x64 tile)
  const int smt0 = wave >> 2,        snt0 = wave & 3;
  const int smt1 = (wave + 8) >> 2,  snt1 = (wave + 8) & 3;

  v8f acc[8] = {};

  for (int kb = 0; kb <= mb; ++kb) {
    const int t0 = kb * 64;

    // warm next key block's K rows and V columns into cache (speculative)
    if (kb < mb) {
      __builtin_prefetch(Kc + (size_t)(t0 + 64 + (lane >> 1)) * 128 + (lane & 1) * 64, 0, 1);
      __builtin_prefetch(Vt + (size_t)(d0 + lane * 4) * L + (t0 + 64), 0, 1);
    }

    // ---- S = Q . Kc^T : batch loads for both subtiles, then issue WMMAs
    v8f s0 = {}, s1 = {};
#pragma unroll
    for (int f0 = 0; f0 < 128; f0 += 32) {
      v16bf a0 = load_frag_a(Q, 128, m0g + smt0 * 16, f0, lane);
      v16bf b0 = load_frag_b(Kc, 128, t0 + snt0 * 16, f0, lane);
      v16bf a1 = load_frag_a(Q, 128, m0g + smt1 * 16, f0, lane);
      v16bf b1 = load_frag_b(Kc, 128, t0 + snt1 * 16, f0, lane);
      s0 = wmma_bf16(a0, b0, s0);
      s1 = wmma_bf16(a1, b1, s1);
    }

    __syncthreads();   // previous P phase finished reading LDS
#pragma unroll
    for (int i = 0; i < 8; ++i) {
      const int lr0 = smt0 * 16 + i + hi;
      const int lc0 = snt0 * 16 + nc;
      float v0 = s0[i];
      if (t0 + lc0 > m0g + lr0) v0 = 0.0f;   // causal: keep t <= l
      Sl[lr0 * 80 + lc0] = (bf16_t)v0;
      const int lr1 = smt1 * 16 + i + hi;
      const int lc1 = snt1 * 16 + nc;
      float v1 = s1[i];
      if (t0 + lc1 > m0g + lr1) v1 = 0.0f;
      Sl[lr1 * 80 + lc1] = (bf16_t)v1;
    }
    __syncthreads();

    // ---- acc += S @ V : load A + all 8 V fragments, then 8 WMMAs
#pragma unroll
    for (int tk = 0; tk < 64; tk += 32) {
      v16bf a = load_frag_a(Sl, 80, mt * 16, tk, lane);
      v16bf vb[8];
#pragma unroll
      for (int j = 0; j < 8; ++j) vb[j] = load_frag_b(Vt, L, d0 + j * 16, t0 + tk, lane);
#pragma unroll
      for (int j = 0; j < 8; ++j) acc[j] = wmma_bf16(a, vb[j], acc[j]);
    }
  }

  // Epilogue: per-row 1/sqrt((l+1)*K) normalization, f32 store.
#pragma unroll
  for (int j = 0; j < 8; ++j) {
    const int n = d0 + j * 16 + nc;
#pragma unroll
    for (int i = 0; i < 8; ++i) {
      const int row = m0g + mt * 16 + i + hi;
      const float norm = rsqrtf((float)(row + 1) * 64.0f);
      Ret[(size_t)row * D + n] = acc[j][i] * norm;
    }
  }
}

// ---------------------------------------------------------------------------
// Row LayerNorm (eps=1e-5) -> bf16.  One block (256 thr) per row, D=512.
// ---------------------------------------------------------------------------
__global__ void layernorm_kernel(const float* __restrict__ Ret, const float* __restrict__ g,
                                 const float* __restrict__ bta, bf16_t* __restrict__ RN, int D) {
  __shared__ float red[256];
  const int row = blockIdx.x;
  const int tid = threadIdx.x;
  const float* r = Ret + (size_t)row * D;
  float v0 = r[tid], v1 = r[tid + 256];
  red[tid] = v0 + v1;
  __syncthreads();
  for (int o = 128; o > 0; o >>= 1) { if (tid < o) red[tid] += red[tid + o]; __syncthreads(); }
  float mean = red[0] / (float)D;
  __syncthreads();
  float d0 = v0 - mean, d1 = v1 - mean;
  red[tid] = d0 * d0 + d1 * d1;
  __syncthreads();
  for (int o = 128; o > 0; o >>= 1) { if (tid < o) red[tid] += red[tid + o]; __syncthreads(); }
  float inv = rsqrtf(red[0] / (float)D + 1e-5f);
  RN[(size_t)row * D + tid]       = (bf16_t)(d0 * inv * g[tid] + bta[tid]);
  RN[(size_t)row * D + tid + 256] = (bf16_t)(d1 * inv * g[tid + 256] + bta[tid + 256]);
}

// ---------------------------------------------------------------------------
// Host-side orchestration
// ---------------------------------------------------------------------------
extern "C" void kernel_launch(void* const* d_in, const int* in_sizes, int n_in,
                              void* d_out, int out_size, void* d_ws, size_t ws_size,
                              hipStream_t stream) {
  (void)in_sizes; (void)n_in; (void)out_size; (void)ws_size;
  constexpr int L = 2048, D = 512, KF = 64;

  const float* x     = (const float*)d_in[0];
  const float* ke_w1 = (const float*)d_in[1];
  const float* ke_b1 = (const float*)d_in[2];
  const float* ke_w2 = (const float*)d_in[3];
  const float* ke_b2 = (const float*)d_in[4];
  const float* qe_w1 = (const float*)d_in[5];
  const float* qe_b1 = (const float*)d_in[6];
  const float* qe_w2 = (const float*)d_in[7];
  const float* qe_b2 = (const float*)d_in[8];
  const float* amp_w = (const float*)d_in[9];
  const float* amp_b = (const float*)d_in[10];
  const float* v_w   = (const float*)d_in[11];
  const float* v_b   = (const float*)d_in[12];
  const float* ln_g  = (const float*)d_in[13];
  const float* ln_b  = (const float*)d_in[14];
  const float* out_w = (const float*)d_in[15];
  const float* out_b = (const float*)d_in[16];

  char* ws = (char*)d_ws;
  size_t off = 0;
  auto alloc = [&](size_t bytes) -> void* {
    void* p = ws + off;
    off = (off + bytes + 255) & ~(size_t)255;
    return p;
  };
  bf16_t* xbf  = (bf16_t*)alloc((size_t)L * D * 2);
  bf16_t* Wke1 = (bf16_t*)alloc((size_t)D * D * 2);
  bf16_t* Wqe1 = (bf16_t*)alloc((size_t)D * D * 2);
  bf16_t* Wv   = (bf16_t*)alloc((size_t)D * D * 2);
  bf16_t* Wout = (bf16_t*)alloc((size_t)D * D * 2);
  bf16_t* Wke2 = (bf16_t*)alloc((size_t)KF * D * 2);
  bf16_t* Wqe2 = (bf16_t*)alloc((size_t)KF * D * 2);
  bf16_t* Wamp = (bf16_t*)alloc((size_t)KF * D * 2);
  bf16_t* Hk   = (bf16_t*)alloc((size_t)L * D * 2);
  bf16_t* Hq   = (bf16_t*)alloc((size_t)L * D * 2);
  float*  ampf = (float*) alloc((size_t)L * KF * 4);
  bf16_t* Qc   = (bf16_t*)alloc((size_t)L * 128 * 2);
  bf16_t* Kc   = (bf16_t*)alloc((size_t)L * 128 * 2);
  bf16_t* Vt   = (bf16_t*)alloc((size_t)D * L * 2);
  float*  Ret  = (float*) alloc((size_t)L * D * 4);
  bf16_t* RN   = (bf16_t*)alloc((size_t)L * D * 2);

  const int thr = 256;
  // --- prep: bf16 convert + weight transposes
  cvt_f32_bf16<<<(L * D + thr - 1) / thr, thr, 0, stream>>>(x, xbf, L * D);
  transpose_f32_bf16<<<(D * D + thr - 1) / thr, thr, 0, stream>>>(ke_w1, Wke1, D, D);
  transpose_f32_bf16<<<(D * D + thr - 1) / thr, thr, 0, stream>>>(qe_w1, Wqe1, D, D);
  transpose_f32_bf16<<<(D * D + thr - 1) / thr, thr, 0, stream>>>(v_w,   Wv,   D, D);
  transpose_f32_bf16<<<(D * D + thr - 1) / thr, thr, 0, stream>>>(out_w, Wout, D, D);
  transpose_f32_bf16<<<(D * KF + thr - 1) / thr, thr, 0, stream>>>(ke_w2, Wke2, D, KF);
  transpose_f32_bf16<<<(D * KF + thr - 1) / thr, thr, 0, stream>>>(qe_w2, Wqe2, D, KF);
  transpose_f32_bf16<<<(D * KF + thr - 1) / thr, thr, 0, stream>>>(amp_w, Wamp, D, KF);

  dim3 blk(256);
  dim3 gD(L / 128, D / 64);   // (16, 8) for N=512 GEMMs
  dim3 gK(L / 128, 1);        // N=64 GEMMs

  // --- encoder hidden layers (GELU fused)
  gemm_kernel<1><<<gD, blk, 0, stream>>>(xbf, Wke1, ke_b1, nullptr, Hk, L, D, D);
  gemm_kernel<1><<<gD, blk, 0, stream>>>(xbf, Wqe1, qe_b1, nullptr, Hq, L, D, D);
  // --- amplitudes (softplus + 0.1 fused)
  gemm_kernel<3><<<gK, blk, 0, stream>>>(xbf, Wamp, amp_b, nullptr, ampf, L, KF, D);
  // --- phasor components packed as [cos | sin] * amp
  phase_kernel<<<dim3(L / 128), blk, 0, stream>>>(Hk, Wke2, ke_b2, ampf, Kc, D);
  phase_kernel<<<dim3(L / 128), blk, 0, stream>>>(Hq, Wqe2, qe_b2, ampf, Qc, D);
  // --- value projection, stored transposed for S@V fragment loads
  gemm_kernel<4><<<gD, blk, 0, stream>>>(xbf, Wv, v_b, nullptr, Vt, L, D, D);
  // --- causal linear attention (the cumsum-phasor core)
  attention_kernel<<<dim3(L / 64, D / 256), blk, 0, stream>>>(Qc, Kc, Vt, Ret, L, D);
  // --- LayerNorm + output projection with x residual
  layernorm_kernel<<<dim3(L), blk, 0, stream>>>(Ret, ln_g, ln_b, RN, D);
  gemm_kernel<2><<<gD, blk, 0, stream>>>(RN, Wout, out_b, x, (float*)d_out, L, D, D);
}
